// Decoder_36051955482725
// MI455X (gfx1250) — matrix operand order repaired and checked by reading, not compiled, not run on hardware
//
#include <hip/hip_runtime.h>
#include <hip/hip_bf16.h>

// ---------------------------------------------------------------------------
// Decoder: embed+BN -> GRU(200) -> GRU(400), Keras reset_after semantics.
// Strategy (MI455X / gfx1250, wave32, WMMA f16 16x16x32):
//  - gx = x@Wx precomputed as wide WMMA GEMMs (time-parallel, whole chip)
//  - recurrence: 8 workgroups x 16 batch rows each; per-step h@Wh via WMMA,
//    weights L2-resident (f16, pre-transposed), f32 state in LDS.
// ---------------------------------------------------------------------------

#define BATCH 128
#define SEQT  512
#define EMBD  200
#define UU1   200
#define UU2   400
#define ROWS  (BATCH * SEQT)        // 65536
#define KP1   224                   // padded K for EMB / U1
#define KP2   416                   // padded K for U2
#define NPG1  208                   // per-gate padded width, layer 1
#define NPG2  400                   // per-gate width, layer 2 (already /16)
#define NC1   (3 * NPG1)            // 624
#define NC2   (3 * NPG2)            // 1200
#define OUT2_ELEMS ((size_t)BATCH * SEQT * UU2)
#define H1_OFF OUT2_ELEMS
#define H2_OFF (OUT2_ELEMS + (size_t)BATCH * UU1)

typedef __attribute__((ext_vector_type(16))) _Float16 v16h;
typedef __attribute__((ext_vector_type(8)))  _Float16 v8h;
typedef __attribute__((ext_vector_type(8)))  float    v8f;

static __device__ __forceinline__ v8f wmma_f16(v16h a, v16h b, v8f c) {
  return __builtin_amdgcn_wmma_f32_16x16x32_f16(
      /*neg_a=*/false, a, /*neg_b=*/false, b,
      /*c_mod=*/(short)0, c, /*reuse_a=*/false, /*reuse_b=*/false);
}

// A fragment (16xK row-major, f16). Lane: m=lane&15, half=lane>>4.
// Element e -> K = kc*32 + half*16 + e (contiguous).
static __device__ __forceinline__ v16h load_a16(const _Float16* base, int ldh,
                                                int kc, int lane) {
  int m = lane & 15, half = lane >> 4;
  const _Float16* p = base + (size_t)m * ldh + kc * 32 + half * 16;
  v8h lo = *(const v8h*)p;
  v8h hi = *(const v8h*)(p + 8);
  v16h a;
#pragma unroll
  for (int i = 0; i < 8; ++i) { a[i] = lo[i]; a[i + 8] = hi[i]; }
  return a;
}

// B fragment from pre-transposed weights Wt[col][Kp] (f16).
// Lane: n=lane&15 (column), element e -> K = kc*32 + (lane>>4)*16 + e (contiguous).
static __device__ __forceinline__ v16h load_bt16(const _Float16* Wt, int Kp,
                                                 int col0, int kc, int lane) {
  int n = lane & 15, kh = lane >> 4;
  const _Float16* p = Wt + (size_t)(col0 + n) * Kp + kc * 32 + kh * 16;
  return *(const v16h*)p;
}

// --------------------------- weight prep -----------------------------------
// W [K][3*U] f32 row-major  ->  Wt [3*Npg][Kp] f16 (transposed, gate-padded)
__global__ void prep_w(const float* __restrict__ W, _Float16* __restrict__ Wt,
                       int K, int U, int Kp, int Npg) {
  size_t idx = (size_t)blockIdx.x * blockDim.x + threadIdx.x;
  size_t total = (size_t)3 * Npg * Kp;
  if (idx >= total) return;
  int k = (int)(idx % Kp);
  int c = (int)(idx / Kp);
  int g = c / Npg, u = c % Npg;
  float v = 0.0f;
  if (u < U && k < K) v = W[(size_t)k * (3 * U) + g * U + u];
  Wt[idx] = (_Float16)v;
}

// --------------------------- embed + BN ------------------------------------
__global__ void embed_bn(const int* __restrict__ inputs, const float* __restrict__ emb,
                         const float* __restrict__ gamma, const float* __restrict__ beta,
                         const float* __restrict__ mean, const float* __restrict__ var,
                         _Float16* __restrict__ xpad) {
  int row = blockIdx.x;          // 0..ROWS-1
  int e = threadIdx.x;           // 0..KP1-1 (blockDim = 224)
  int tok = inputs[row];
  float y = 0.0f;
  if (e < EMBD) {
    float x = emb[(size_t)tok * EMBD + e];
    y = (x - mean[e]) * rsqrtf(var[e] + 1.0e-3f) * gamma[e] + beta[e];
  }
  xpad[(size_t)row * KP1 + e] = (_Float16)y;
}

// --------------------------- WMMA GEMM -------------------------------------
// C[ROWS][Ncols] f32 = A[ROWS][Kp] f16 @ Bt^T + bias (per-gate padded bias).
// grid = (Ncols/16, ROWS/16/8), block = 256 (8 waves, one M-tile each).
__global__ void gemm_wmma(const _Float16* __restrict__ A, const _Float16* __restrict__ Bt,
                          const float* __restrict__ bias, float* __restrict__ C,
                          int Kp, int Ncols, int Npg, int Ureal) {
  int wave = threadIdx.x >> 5;
  int lane = threadIdx.x & 31;
  int ntile = blockIdx.x;
  int mtile = blockIdx.y * 8 + wave;
  const _Float16* Arow = A + (size_t)mtile * 16 * Kp;
  v8f acc = {};
  int kchunks = Kp >> 5;
  for (int kc = 0; kc < kchunks; ++kc) {
    v16h a = load_a16(Arow, Kp, kc, lane);
    v16h b = load_bt16(Bt, Kp, ntile * 16, kc, lane);
    acc = wmma_f16(a, b, acc);
  }
  int c = ntile * 16 + (lane & 15);
  int g = c / Npg, u = c % Npg;
  float bv = (u < Ureal) ? bias[g * Ureal + u] : 0.0f;
  int mbase = mtile * 16 + (lane >> 4) * 8;
#pragma unroll
  for (int e = 0; e < 8; ++e)
    C[(size_t)(mbase + e) * Ncols + c] = acc[e] + bv;
}

// --------------------------- GRU recurrence --------------------------------
// One block = 16 batch rows, blockDim = (NPG/16) waves * 32.
// Each wave owns one 16-col group per gate: tiles jz, jz+NG, jz+2*NG.
template <int U, int KP, int NPG, bool STORE_F16>
__global__ void gru_rec(const float* __restrict__ gx,       // [ROWS][3*NPG] f32 (incl. bx)
                        const _Float16* __restrict__ Wht,   // [3*NPG][KP] f16
                        const float* __restrict__ bh,       // [3*U] f32
                        const int* __restrict__ inputs,     // [B][T]
                        const float* __restrict__ h0,       // [B][U]
                        _Float16* __restrict__ out16,       // layer1: padded outputs [ROWS][KP1]
                        float* __restrict__ out32,          // layer2: d_out out2 [B][T][U]
                        float* __restrict__ hfin) {         // [B][U]
  constexpr int NG = NPG / 16;
  constexpr int KC = KP / 32;
  constexpr int NCOLS = 3 * NPG;
  __shared__ float h32[16 * NPG];
  __shared__ float prevo[16 * NPG];
  __shared__ __align__(32) _Float16 h16[16 * KP];

  const int wave = threadIdx.x >> 5;
  const int lane = threadIdx.x & 31;
  const int bb0 = blockIdx.x * 16;

  for (int i = threadIdx.x; i < 16 * NPG; i += blockDim.x) {
    int m = i / NPG, u = i % NPG;
    float h = (u < U) ? h0[(size_t)(bb0 + m) * U + u] : 0.0f;
    h32[i] = h;
    prevo[i] = 0.0f;
  }
  for (int i = threadIdx.x; i < 16 * KP; i += blockDim.x) {
    int m = i / KP, k = i % KP;
    h16[i] = (_Float16)((k < U) ? h32[m * NPG + k] : 0.0f);
  }
  __syncthreads();

  const int jz = wave;                 // column group
  const int n = lane & 15, hi = lane >> 4;
  const int uz = jz * 16 + n;
  const float bhz = (uz < U) ? bh[0 * U + uz] : 0.0f;
  const float bhr = (uz < U) ? bh[1 * U + uz] : 0.0f;
  const float bhh = (uz < U) ? bh[2 * U + uz] : 0.0f;

  for (int t = 0; t < SEQT; ++t) {
    v8f accZ = {}, accR = {}, accH = {};
#pragma unroll
    for (int kc = 0; kc < KC; ++kc) {
      v16h a = load_a16(h16, KP, kc, lane);
      v16h bz = load_bt16(Wht, KP, 0 * NPG + jz * 16, kc, lane);
      accZ = wmma_f16(a, bz, accZ);
      v16h br = load_bt16(Wht, KP, 1 * NPG + jz * 16, kc, lane);
      accR = wmma_f16(a, br, accR);
      v16h bb = load_bt16(Wht, KP, 2 * NPG + jz * 16, kc, lane);
      accH = wmma_f16(a, bb, accH);
    }
    __syncthreads();  // all h16 reads complete before state update

#pragma unroll
    for (int e = 0; e < 8; ++e) {
      int m = hi * 8 + e;
      int bbi = bb0 + m;
      size_t grow = ((size_t)bbi * SEQT + t) * NCOLS;
      float gz = gx[grow + 0 * NPG + uz];
      float gr = gx[grow + 1 * NPG + uz];
      float gh = gx[grow + 2 * NPG + uz];
      float z = 1.0f / (1.0f + expf(-(gz + accZ[e] + bhz)));
      float r = 1.0f / (1.0f + expf(-(gr + accR[e] + bhr)));
      float hc = tanhf(gh + r * (accH[e] + bhh));
      float hold = h32[m * NPG + uz];
      float hnew = z * hold + (1.0f - z) * hc;
      bool msk = inputs[(size_t)bbi * SEQT + t] != 0;
      if (!msk) hnew = hold;
      float o = msk ? hnew : prevo[m * NPG + uz];
      h32[m * NPG + uz] = hnew;
      prevo[m * NPG + uz] = o;
      h16[m * KP + uz] = (_Float16)hnew;
      if (STORE_F16) {
        out16[((size_t)bbi * SEQT + t) * KP1 + uz] = (_Float16)o;
      } else {
        out32[((size_t)bbi * SEQT + t) * U + uz] = o;  // uz < 400 == U always
      }
    }
    __syncthreads();  // state visible before next step's A loads
  }

  if (uz < U) {
#pragma unroll
    for (int e = 0; e < 8; ++e) {
      int m = hi * 8 + e;
      hfin[(size_t)(bb0 + m) * U + uz] = h32[m * NPG + uz];
    }
  }
}

// --------------------------- host launcher ---------------------------------
extern "C" void kernel_launch(void* const* d_in, const int* in_sizes, int n_in,
                              void* d_out, int out_size, void* d_ws, size_t ws_size,
                              hipStream_t stream) {
  (void)in_sizes; (void)n_in; (void)out_size; (void)ws_size;
  const int*   inputs = (const int*)d_in[0];
  const float* state0 = (const float*)d_in[1];
  const float* state1 = (const float*)d_in[2];
  // d_in[3] = encoder_outputs (unused by reference)
  const float* emb    = (const float*)d_in[4];
  const float* gamma  = (const float*)d_in[5];
  const float* beta   = (const float*)d_in[6];
  const float* mean   = (const float*)d_in[7];
  const float* var    = (const float*)d_in[8];
  const float* Wx1    = (const float*)d_in[9];
  const float* Wh1    = (const float*)d_in[10];
  const float* b1     = (const float*)d_in[11];   // [2][600]
  const float* Wx2    = (const float*)d_in[12];
  const float* Wh2    = (const float*)d_in[13];
  const float* b2     = (const float*)d_in[14];   // [2][1200]
  float* out = (float*)d_out;

  char* ws = (char*)d_ws;
  size_t off = 0;
  auto carve = [&](size_t bytes) -> void* {
    void* p = ws + off;
    off = (off + bytes + 255) & ~(size_t)255;
    return p;
  };
  _Float16* xpad    = (_Float16*)carve((size_t)ROWS * KP1 * 2);   // 29.4 MB
  _Float16* out1pad = (_Float16*)carve((size_t)ROWS * KP1 * 2);   // 29.4 MB
  _Float16* Wx1t    = (_Float16*)carve((size_t)NC1 * KP1 * 2);
  _Float16* Wh1t    = (_Float16*)carve((size_t)NC1 * KP1 * 2);
  _Float16* Wx2t    = (_Float16*)carve((size_t)NC2 * KP1 * 2);
  _Float16* Wh2t    = (_Float16*)carve((size_t)NC2 * KP2 * 2);
  float*    gx1     = (float*)carve((size_t)ROWS * NC1 * 4);      // 164 MB
  float*    gx2     = (float*)carve((size_t)ROWS * NC2 * 4);      // 315 MB

  // 1) weight transpose/pad/convert
  {
    auto blocks = [](size_t total) { return (unsigned)((total + 255) / 256); };
    prep_w<<<blocks((size_t)NC1 * KP1), 256, 0, stream>>>(Wx1, Wx1t, EMBD, UU1, KP1, NPG1);
    prep_w<<<blocks((size_t)NC1 * KP1), 256, 0, stream>>>(Wh1, Wh1t, UU1,  UU1, KP1, NPG1);
    prep_w<<<blocks((size_t)NC2 * KP1), 256, 0, stream>>>(Wx2, Wx2t, UU1,  UU2, KP1, NPG2);
    prep_w<<<blocks((size_t)NC2 * KP2), 256, 0, stream>>>(Wh2, Wh2t, UU2,  UU2, KP2, NPG2);
  }

  // 2) embedding + batchnorm -> padded f16 activations
  embed_bn<<<ROWS, KP1, 0, stream>>>(inputs, emb, gamma, beta, mean, var, xpad);

  // 3) gx1 = x @ Wx1 + bx1   (time-parallel WMMA GEMM)
  gemm_wmma<<<dim3(NC1 / 16, ROWS / 16 / 8), 256, 0, stream>>>(
      xpad, Wx1t, b1, gx1, KP1, NC1, NPG1, UU1);

  // 4) layer-1 recurrence (8 blocks x 13 waves); zero padded out1 cols first
  hipMemsetAsync(out1pad, 0, (size_t)ROWS * KP1 * 2, stream);
  gru_rec<UU1, KP1, NPG1, true><<<BATCH / 16, (NPG1 / 16) * 32, 0, stream>>>(
      gx1, Wh1t, b1 + 3 * UU1, inputs, state0, out1pad, nullptr, out + H1_OFF);

  // 5) gx2 = out1 @ Wx2 + bx2
  gemm_wmma<<<dim3(NC2 / 16, ROWS / 16 / 8), 256, 0, stream>>>(
      out1pad, Wx2t, b2, gx2, KP1, NC2, NPG2, UU2);

  // 6) layer-2 recurrence (8 blocks x 25 waves); writes out2 + h2
  gru_rec<UU2, KP2, NPG2, false><<<BATCH / 16, (NPG2 / 16) * 32, 0, stream>>>(
      gx2, Wh2t, b2 + 3 * UU2, inputs, state1, nullptr, out, out + H2_OFF);
}
